// Self_Attention_58686433133024
// MI455X (gfx1250) — matrix operand order repaired and checked by reading, not compiled, not run on hardware
//
#include <hip/hip_runtime.h>

#define BATCH 2
#define SEQ   4096
#define EMB   512

typedef _Float16 half_t;
typedef __attribute__((ext_vector_type(16))) _Float16 v16h;
typedef __attribute__((ext_vector_type(8)))  _Float16 v8h;
typedef __attribute__((ext_vector_type(8)))  float    v8f;
typedef __attribute__((ext_vector_type(4)))  int      v4i;

#if __has_builtin(__builtin_amdgcn_global_load_async_to_lds_b128) && \
    __has_builtin(__builtin_amdgcn_s_wait_asynccnt)
#define USE_ASYNC 1
#else
#define USE_ASYNC 0
#endif

// 16-byte global -> LDS copy: async (ASYNCcnt-tracked) when available.
static __device__ __forceinline__ void copy16(const half_t* g, half_t* l) {
#if USE_ASYNC
  __builtin_amdgcn_global_load_async_to_lds_b128((v4i*)g, (v4i*)l, 0, 0);
#else
  *(uint4*)l = *(const uint4*)g;
#endif
}
// Wait until at most N async copies are outstanding (in-order completion).
template <int N>
static __device__ __forceinline__ void async_wait() {
#if USE_ASYNC
  __builtin_amdgcn_s_wait_asynccnt(N);
#endif
}

// D = A(16x32 f16) * B(32x16 f16) + C(16x16 f32)   -> v_wmma_f32_16x16x32_f16
static __device__ __forceinline__ v8f wmma_f16(v16h a, v16h b, v8f c) {
  return __builtin_amdgcn_wmma_f32_16x16x32_f16(false, a, false, b, (short)0, c, false, false);
}

static __device__ __forceinline__ v16h combine8(v8h lo, v8h hi) {
  return __builtin_shufflevector(lo, hi, 0, 1, 2, 3, 4, 5, 6, 7, 8, 9, 10, 11, 12, 13, 14, 15);
}

// A fragment (16x32 f16) from [m][k] storage, k unit-stride: 2x ds_load_b128
static __device__ __forceinline__ v16h load_frag_a(const half_t* p, int strideM, int lane) {
  const int laneM = lane & 15, h = lane >> 4;
  const half_t* q = p + laneM * strideM + h * 8;
  v8h lo = *(const v8h*)q;
  v8h hi = *(const v8h*)(q + 16);
  return combine8(lo, hi);
}

// B fragment (32x16 f16) from [n][k] storage, k unit-stride: 2x ds_load_b128
static __device__ __forceinline__ v16h load_frag_b(const half_t* p, int strideN, int lane) {
  const int laneN = lane & 15, h = lane >> 4;
  const half_t* q = p + laneN * strideN + h * 8;
  v8h lo = *(const v8h*)q;
  v8h hi = *(const v8h*)(q + 16);
  return combine8(lo, hi);
}

// Convert 8 floats (2x float4) to one 16B v8h store.
static __device__ __forceinline__ void cvt8_store(const float4* s4, half_t* dst) {
  const float4 f0 = s4[0], f1 = s4[1];
  v8h hv = {(half_t)f0.x, (half_t)f0.y, (half_t)f0.z, (half_t)f0.w,
            (half_t)f1.x, (half_t)f1.y, (half_t)f1.z, (half_t)f1.w};
  *(v8h*)dst = hv;
}

// ---------------------------------------------------------------------------
// GEMM: C[m,n] = sum_e X[m,e] * W[n,e]  (+bias) ; K = N = EMB = 512.
// Block: 64x64 tile, 8 waves, each wave 16 rows x 32 cols.
// Bs stored [n][k] (straight copy of W rows). OUT_TR: out[b][n][s] (V^T).
// ---------------------------------------------------------------------------
template <typename TIn, bool ADD_BIAS, bool OUT_HALF, bool OUT_TR>
__global__ void __launch_bounds__(256)
gemm_xwt_kernel(const TIn* __restrict__ X, const float* __restrict__ W,
                const float* __restrict__ bias, half_t* __restrict__ outH,
                float* __restrict__ outF) {
  constexpr int LSTR = 72;  // padded LDS stride (halves); rows 16B-aligned
  __shared__ half_t As[64 * LSTR];
  __shared__ half_t Bs[64 * LSTR];

  const int m0 = blockIdx.x * 64;
  const int n0 = blockIdx.y * 64;
  const int tid = threadIdx.x;
  const int wave = tid >> 5, lane = tid & 31;
  const int rowg = wave >> 1;        // 0..3 : 16-row group
  const int colp = (wave & 1) * 2;   // 0 or 2 : pair of 16-col subtiles
  const int laneN = lane & 15, lh = lane >> 4;

  v8f acc0 = {}, acc1 = {};

  for (int k0 = 0; k0 < EMB; k0 += 64) {
    {  // stage A [m][k] (16 elems/thread)
      const int r = tid >> 2, q = tid & 3;
      half_t* dst = &As[r * LSTR + q * 16];
      if constexpr (sizeof(TIn) == 4) {
        const float4* s4 = (const float4*)(X + (size_t)(m0 + r) * EMB + k0 + q * 16);
        cvt8_store(s4, dst);
        cvt8_store(s4 + 2, dst + 8);
      } else {
        const half_t* src = (const half_t*)X + (size_t)(m0 + r) * EMB + k0 + q * 16;
        copy16(src, dst);
        copy16(src + 8, dst + 8);
      }
    }
    {  // stage B [n][k]: Bs[f][e] = W[n0+f][k0+e]
      const int f = tid >> 2, q = tid & 3;
      const float4* s4 = (const float4*)(W + (size_t)(n0 + f) * EMB + k0 + q * 16);
      half_t* dst = &Bs[f * LSTR + q * 16];
      cvt8_store(s4, dst);
      cvt8_store(s4 + 2, dst + 8);
    }
    async_wait<0>();
    __syncthreads();
#pragma unroll
    for (int kk = 0; kk < 64; kk += 32) {
      v16h a  = load_frag_a(&As[rowg * 16 * LSTR + kk], LSTR, lane);
      v16h b0 = load_frag_b(&Bs[(colp    ) * 16 * LSTR + kk], LSTR, lane);
      v16h b1 = load_frag_b(&Bs[(colp + 1) * 16 * LSTR + kk], LSTR, lane);
      acc0 = wmma_f16(a, b0, acc0);
      acc1 = wmma_f16(a, b1, acc1);
    }
    __syncthreads();
  }

#pragma unroll
  for (int r = 0; r < 8; ++r) {
    const int row = m0 + rowg * 16 + r + 8 * lh;
    const int c0  = n0 + (colp    ) * 16 + laneN;
    const int c1  = n0 + (colp + 1) * 16 + laneN;
    float v0 = acc0[r], v1 = acc1[r];
    if constexpr (ADD_BIAS) { v0 += bias[c0]; v1 += bias[c1]; }
    if constexpr (OUT_TR) {  // out[b][n][s] : V^T layout per batch
      const int b = row >> 12, s = row & (SEQ - 1);
      outH[(size_t)b * EMB * SEQ + (size_t)c0 * SEQ + s] = (half_t)v0;
      outH[(size_t)b * EMB * SEQ + (size_t)c1 * SEQ + s] = (half_t)v1;
    } else if constexpr (OUT_HALF) {
      outH[(size_t)row * EMB + c0] = (half_t)v0;
      outH[(size_t)row * EMB + c1] = (half_t)v1;
    } else {
      outF[(size_t)row * EMB + c0] = v0;
      outF[(size_t)row * EMB + c1] = v1;
    }
  }
}

// ---------------------------------------------------------------------------
// Flash attention, anti-causal mask (keep j >= i).
// Block: 32 query rows x d=512; 64-key tiles from the diagonal forward.
// ASYNCcnt software pipeline (in-order completion):
//   per tile, per thread: 16 K copies then 16 V^T copies;
//   wait<16> before QK  => K tile done, V^T still in flight;
//   K(j+1) prefetched into the other K buffer during softmax/PV;
//   wait<16> before PV  => V^T(j) done, K(j+1) still in flight.
// WMMA loops software-pipeline their B-fragment ds_load_b128 pairs.
// LDS: Q[32][520]  K2x[64][520]  Vt[512][72]  P[32][72]  S[32][72]f  (~255 KB)
// ---------------------------------------------------------------------------
__global__ void __launch_bounds__(256)
attn_kernel(const half_t* __restrict__ Qg, const half_t* __restrict__ Kg,
            const half_t* __restrict__ Vtg, half_t* __restrict__ Og) {
  extern __shared__ char smem[];
  half_t* Qs  = (half_t*)smem;            // 32 x 520
  half_t* Ks0 = Qs + 32 * 520;            // 64 x 520  (ping)
  half_t* Ks1 = Ks0 + 64 * 520;           // 64 x 520  (pong)
  half_t* Vt  = Ks1 + 64 * 520;           // 512 x 72  ([e][key])
  half_t* Ps  = Vt + 512 * 72;            // 32 x 72
  float*  Ss  = (float*)(Ps + 32 * 72);   // 32 x 72
  float*  red = Ss + 32 * 72;             // 256
  float*  mrow = red + 256;               // 32
  float*  lrow = mrow + 32;               // 32
  float*  arow = lrow + 32;               // 32

  const int q0 = blockIdx.x * 32;
  const size_t bo = (size_t)blockIdx.y * SEQ * EMB;
  const int tid = threadIdx.x;
  const int wave = tid >> 5, lane = tid & 31;
  const int rowg = wave & 1;
  const int colg = wave >> 1;
  const int laneN = lane & 15, lh = lane >> 4;
  const float inv_scale = 0.04419417382415922f;  // 1/sqrt(512)

  // stage K tile [key][e]: 64 rows x 64 x 16B (16 copies/thread)
  auto stage_k = [&](int j0, half_t* dstK) {
    for (int idx = tid; idx < 64 * (EMB / 8); idx += 256) {
      const int r = idx >> 6, c8 = idx & 63;
      copy16(Kg + bo + (size_t)(j0 + r) * EMB + c8 * 8, &dstK[r * 520 + c8 * 8]);
    }
  };

  // stage Q tile: 32 rows x 64 x 16B
  for (int idx = tid; idx < 32 * (EMB / 8); idx += 256) {
    const int r = idx >> 6, c8 = idx & 63;
    copy16(Qg + bo + (size_t)(q0 + r) * EMB + c8 * 8, &Qs[r * 520 + c8 * 8]);
  }
  if (tid < 32) { mrow[tid] = -1e30f; lrow[tid] = 0.0f; arow[tid] = 1.0f; }
  async_wait<0>();
  __syncthreads();

  const int jstart = (q0 / 64) * 64;
  stage_k(jstart, Ks0);  // prologue K prefetch, overlaps with Q-fragment hoist

  // hoist loop-invariant Q fragments (16 x 8 VGPRs)
  v16h qf[16];
#pragma unroll
  for (int t = 0; t < 16; ++t)
    qf[t] = load_frag_a(&Qs[(rowg * 16) * 520 + t * 32], 520, lane);

  v8f o[8];
#pragma unroll
  for (int s = 0; s < 8; ++s) o[s] = v8f{};

  const int sr = tid & 31;   // softmax row
  const int sg = tid >> 5;   // softmax 8-col group

  int it = 0;
  for (int j0 = jstart; j0 < SEQ; j0 += 64, ++it) {
    half_t* Kcur = (it & 1) ? Ks1 : Ks0;
    half_t* Knxt = (it & 1) ? Ks0 : Ks1;
    const bool havenext = (j0 + 64) < SEQ;

    __syncthreads();  // prev iteration's Vt/Ss/Ps reads complete
    // stage V^T tile [e][key]: 512 rows x 8 x 16B (16 copies/thread)
    for (int idx = tid; idx < EMB * 8; idx += 256) {
      const int r = idx >> 3, c8 = idx & 7;
      copy16(Vtg + bo + (size_t)r * SEQ + j0 + c8 * 8, &Vt[r * 72 + c8 * 8]);
    }
    async_wait<16>();  // oldest 16/thread (= K(j0) copies) complete
    __syncthreads();

    // S(32x64) = Q * K^T : 16 WMMAs over e, B-frag loads pipelined 1 ahead
    v8f sacc = {};
    {
      const half_t* kb = &Kcur[(colg * 16) * 520];
      v16h b0 = load_frag_b(kb, 520, lane);
#pragma unroll
      for (int t = 0; t < 15; ++t) {
        v16h b1 = load_frag_b(kb + (t + 1) * 32, 520, lane);
        sacc = wmma_f16(qf[t], b0, sacc);
        b0 = b1;
      }
      sacc = wmma_f16(qf[15], b0, sacc);
    }

    // prefetch next K tile into the other buffer (overlaps softmax + PV)
    if (havenext) stage_k(j0 + 64, Knxt);

#pragma unroll
    for (int r = 0; r < 8; ++r) {
      const int li = rowg * 16 + r + 8 * lh;
      const int lj = colg * 16 + laneN;
      float sv = sacc[r] * inv_scale;
      if ((j0 + lj) < (q0 + li)) sv = -1e30f;  // anti-causal: keep j >= i
      Ss[li * 72 + lj] = sv;
    }
    __syncthreads();

    // online softmax, parallel over 256 threads (row = tid&31, 8 cols each)
    {
      float pm = -1e30f;
#pragma unroll
      for (int c = 0; c < 8; ++c) pm = fmaxf(pm, Ss[sr * 72 + sg * 8 + c]);
      red[sr * 8 + sg] = pm;
    }
    __syncthreads();
    if (tid < 32) {
      float mx = mrow[tid];
#pragma unroll
      for (int g = 0; g < 8; ++g) mx = fmaxf(mx, red[tid * 8 + g]);
      arow[tid] = __expf(mrow[tid] - mx);
      mrow[tid] = mx;
    }
    __syncthreads();
    {
      const float mx = mrow[sr];
      float ps = 0.0f;
#pragma unroll
      for (int c = 0; c < 8; ++c) {
        const float p = __expf(Ss[sr * 72 + sg * 8 + c] - mx);
        ps += p;
        Ps[sr * 72 + sg * 8 + c] = (half_t)p;
      }
      red[sr * 8 + sg] = ps;
    }
    __syncthreads();
    if (tid < 32) {
      float sum = 0.0f;
#pragma unroll
      for (int g = 0; g < 8; ++g) sum += red[tid * 8 + g];
      lrow[tid] = lrow[tid] * arow[tid] + sum;
    }
    // V^T(j0) done: with prefetch in flight the newest 16 may remain outstanding
    if (havenext) async_wait<16>(); else async_wait<0>();
    __syncthreads();

    // rescale O, then O += P(32x64) * V(64x512); wave: 16 rows x 128 cols
    float alv[8];
#pragma unroll
    for (int r = 0; r < 8; ++r) alv[r] = arow[rowg * 16 + r + 8 * lh];
#pragma unroll
    for (int s = 0; s < 8; ++s)
#pragma unroll
      for (int r = 0; r < 8; ++r) o[s][r] = o[s][r] * alv[r];

#pragma unroll
    for (int kk = 0; kk < 64; kk += 32) {
      v16h a = load_frag_a(&Ps[(rowg * 16) * 72 + kk], 72, lane);
      const half_t* vb = &Vt[(colg * 128) * 72 + kk];
      v16h b0 = load_frag_b(vb, 72, lane);
#pragma unroll
      for (int s = 0; s < 7; ++s) {
        v16h b1 = load_frag_b(vb + (s + 1) * 16 * 72, 72, lane);
        o[s] = wmma_f16(a, b0, o[s]);
        b0 = b1;
      }
      o[7] = wmma_f16(a, b0, o[7]);
    }
  }

  // finalize: divide by row sum, store f16 O to workspace
  float linv[8];
#pragma unroll
  for (int r = 0; r < 8; ++r) linv[r] = 1.0f / lrow[rowg * 16 + r + 8 * lh];
#pragma unroll
  for (int s = 0; s < 8; ++s)
#pragma unroll
    for (int r = 0; r < 8; ++r) {
      const int row = q0 + rowg * 16 + r + 8 * lh;
      const int col = colg * 128 + s * 16 + laneN;
      Og[bo + (size_t)row * EMB + col] = (half_t)(o[s][r] * linv[r]);
    }
}

// ---------------------------------------------------------------------------
extern "C" void kernel_launch(void* const* d_in, const int* in_sizes, int n_in,
                              void* d_out, int out_size, void* d_ws, size_t ws_size,
                              hipStream_t stream) {
  const float* x  = (const float*)d_in[0];
  const float* Wq = (const float*)d_in[1];
  const float* Wk = (const float*)d_in[2];
  const float* Wv = (const float*)d_in[3];
  const float* Wo = (const float*)d_in[4];
  const float* bo = (const float*)d_in[5];
  float* out = (float*)d_out;

  const size_t tokens = (size_t)BATCH * SEQ;  // 8192
  const size_t plane  = tokens * EMB;         // 4,194,304 elems
  half_t* Qw  = (half_t*)d_ws;
  half_t* Kw  = Qw + plane;
  half_t* Vtw = Kw + plane;                   // V stored transposed [b][e][s]
  half_t* Ow  = Vtw + plane;                  // 32 MB total f16

  dim3 gproj((unsigned)(tokens / 64), EMB / 64);  // (128, 8)
  gemm_xwt_kernel<float, false, true, false><<<gproj, 256, 0, stream>>>(x, Wq, nullptr, Qw, nullptr);
  gemm_xwt_kernel<float, false, true, false><<<gproj, 256, 0, stream>>>(x, Wk, nullptr, Kw, nullptr);
  gemm_xwt_kernel<float, false, true, true ><<<gproj, 256, 0, stream>>>(x, Wv, nullptr, Vtw, nullptr);

  const size_t smem = (size_t)(32 * 520 + 2 * 64 * 520 + 512 * 72 + 32 * 72) * sizeof(half_t)
                    + (size_t)(32 * 72 + 256 + 96) * sizeof(float);  // ~255 KB (< 320 KB WGP LDS)
  (void)hipFuncSetAttribute((const void*)attn_kernel,
                            hipFuncAttributeMaxDynamicSharedMemorySize, (int)smem);
  dim3 gattn(SEQ / 32, BATCH);  // (128, 2)
  attn_kernel<<<gattn, 256, smem, stream>>>(Qw, Kw, Vtw, Ow);

  gemm_xwt_kernel<half_t, true, false, false><<<gproj, 256, 0, stream>>>(Ow, Wo, bo, nullptr, out);
}
